// Baller2Vec_22058952032834
// MI455X (gfx1250) — compile-verified, matrix operand order checked by reading, not run on hardware
//
#include <hip/hip_runtime.h>

typedef _Float16 half_t;
typedef __attribute__((ext_vector_type(16))) _Float16 v16h;
typedef __attribute__((ext_vector_type(8)))  _Float16 v8h;
typedef __attribute__((ext_vector_type(8)))  float    v8f;

#define WMMA_F16(a,b,c) __builtin_amdgcn_wmma_f32_16x16x32_f16(false,(a),false,(b),(short)0,(c),false,false)

#define NTOK 3600
#define DMODEL 512
#define VT_LD 3616   // padded token stride for transposed V
#define SQRT_D 22.62741699796952f

__device__ __forceinline__ int stepOf(int i){
  return (i < 3000) ? (i / 10) : ((i < 3300) ? (i - 3000) : (i - 3300));
}

__device__ __forceinline__ v16h ld16(const half_t* __restrict__ p0,
                                     const half_t* __restrict__ p1){
  v8h a = *(const v8h*)p0;
  v8h b = *(const v8h*)p1;
  v16h r;
#pragma unroll
  for (int i = 0; i < 8; ++i){ r[i] = a[i]; r[i+8] = b[i]; }
  return r;
}

// ---------------------------------------------------------------- f32 -> f16
__global__ void k_cvt(const float* __restrict__ s, half_t* __restrict__ d, int n){
  int i = blockIdx.x * 256 + threadIdx.x;
  if (i < n) d[i] = (half_t)s[i];
}

// f32 [rows x ks] -> f16 [rows x kd], zero-padded columns
__global__ void k_cvt_pad(const float* __restrict__ s, half_t* __restrict__ d,
                          int rows, int ks, int kd){
  int i = blockIdx.x * 256 + threadIdx.x;
  if (i >= rows * kd) return;
  int r = i / kd, c = i - r * kd;
  d[i] = (half_t)(c < ks ? s[r * ks + c] : 0.f);
}

// ---------------------------------------------------------------- build padded MLP input [3300 x 32]
__global__ void k_build_in(
    const float* __restrict__ pxs, const float* __restrict__ pys, const float* __restrict__ phs,
    const float* __restrict__ bxs, const float* __restrict__ bys, const float* __restrict__ bzs,
    const float* __restrict__ emb, const float* __restrict__ bemb,
    const int* __restrict__ pidx, half_t* __restrict__ inp)
{
  int i = blockIdx.x * 256 + threadIdx.x;
  if (i >= 3300 * 32) return;
  int tok = i >> 5, c = i & 31;
  float v = 0.f;
  if (tok < 3000){
    if (c < 20)       v = emb[(size_t)pidx[tok] * 20 + c];
    else if (c == 20) v = pxs[tok];
    else if (c == 21) v = pys[tok];
    else if (c == 22) v = phs[tok];
  } else {
    int t = tok - 3000;
    if (c < 20)       v = bemb[c];
    else if (c == 20) v = bxs[t];
    else if (c == 21) v = bys[t];
    else if (c == 22) v = bzs[t];
  }
  inp[i] = (half_t)v;
}

// CLS rows 3300..3599 = cls_emb
__global__ void k_cls(const float* __restrict__ cls, float* __restrict__ x, half_t* __restrict__ xh){
  int i = blockIdx.x * 256 + threadIdx.x;
  if (i >= 300 * DMODEL) return;
  int r = 3300 + (i >> 9), c = i & 511;
  float v = cls[c];
  x [(size_t)r * DMODEL + c] = v;
  xh[(size_t)r * DMODEL + c] = (half_t)v;
}

// ---------------------------------------------------------------- WMMA GEMM
// C[M,N] = (A[M,K](f16) * B[N,K](f16)^T + bias) * scale, optional relu.
// Wave tile 32x64: 2 A-frags x 4 B-frags -> 8 WMMAs per K-step, double-buffered.
// Lane offsets folded into 6 pointers advanced by 64B/step (no per-iter addr math).
// launch_bounds(256,1): ~180 VGPRs live, must not spill.
__global__ __launch_bounds__(256, 1)
void k_gemm(const half_t* __restrict__ A, const half_t* __restrict__ B,
            const float* __restrict__ bias,
            float* __restrict__ C, half_t* __restrict__ Ch,
            int M, int N, int K, int relu, float scale)
{
  int wave   = threadIdx.x >> 5;
  int lane   = threadIdx.x & 31;
  int mbase  = blockIdx.y * 128 + (wave >> 1) * 32;  // 4 row groups x 32 rows
  int nbase0 = blockIdx.x * 128 + (wave &  1) * 64;  // 2 col groups x 64 cols
  if (mbase >= M || nbase0 >= N) return;

  int lrow = lane & 15;
  int khf  = (lane >> 4) * 8;

  int m1 = (mbase + 16 < M) ? (mbase + 16) : mbase;  // second row-half (dup if OOB)
  int ra0 = M - 1 - mbase; if (ra0 > 15) ra0 = 15;
  int ra1 = M - 1 - m1;    if (ra1 > 15) ra1 = 15;
  int r0 = lrow > ra0 ? ra0 : lrow;
  int r1 = lrow > ra1 ? ra1 : lrow;
  const half_t* pA0 = A + (size_t)(mbase + r0) * K + khf;
  const half_t* pA1 = A + (size_t)(m1    + r1) * K + khf;

  const half_t* pBt[4];
#pragma unroll
  for (int t = 0; t < 4; ++t){
    int nb = nbase0 + t * 16;
    int nbc = nb < N ? nb : 0;          // inactive tiles read tile 0 (discarded)
    int rr = N - 1 - nbc; if (rr > 15) rr = 15;
    int rbl = lrow > rr ? rr : lrow;
    pBt[t] = B + (size_t)(nbc + rbl) * K + khf;
  }
  const half_t* pB0 = pBt[0]; const half_t* pB1 = pBt[1];
  const half_t* pB2 = pBt[2]; const half_t* pB3 = pBt[3];

  v8f acc[8] = {};
  v16h a0 = ld16(pA0, pA0 + 16); pA0 += 32;
  v16h a1 = ld16(pA1, pA1 + 16); pA1 += 32;
  v16h b0 = ld16(pB0, pB0 + 16); pB0 += 32;
  v16h b1 = ld16(pB1, pB1 + 16); pB1 += 32;
  v16h b2 = ld16(pB2, pB2 + 16); pB2 += 32;
  v16h b3 = ld16(pB3, pB3 + 16); pB3 += 32;

  for (int k = 32; k < K; k += 32){
    v16h na0 = ld16(pA0, pA0 + 16); pA0 += 32;
    v16h na1 = ld16(pA1, pA1 + 16); pA1 += 32;
    v16h nb0 = ld16(pB0, pB0 + 16); pB0 += 32;
    v16h nb1 = ld16(pB1, pB1 + 16); pB1 += 32;
    v16h nb2 = ld16(pB2, pB2 + 16); pB2 += 32;
    v16h nb3 = ld16(pB3, pB3 + 16); pB3 += 32;
    acc[0] = WMMA_F16(a0, b0, acc[0]);
    acc[1] = WMMA_F16(a0, b1, acc[1]);
    acc[2] = WMMA_F16(a0, b2, acc[2]);
    acc[3] = WMMA_F16(a0, b3, acc[3]);
    acc[4] = WMMA_F16(a1, b0, acc[4]);
    acc[5] = WMMA_F16(a1, b1, acc[5]);
    acc[6] = WMMA_F16(a1, b2, acc[6]);
    acc[7] = WMMA_F16(a1, b3, acc[7]);
    a0 = na0; a1 = na1; b0 = nb0; b1 = nb1; b2 = nb2; b3 = nb3;
  }
  acc[0] = WMMA_F16(a0, b0, acc[0]);
  acc[1] = WMMA_F16(a0, b1, acc[1]);
  acc[2] = WMMA_F16(a0, b2, acc[2]);
  acc[3] = WMMA_F16(a0, b3, acc[3]);
  acc[4] = WMMA_F16(a1, b0, acc[4]);
  acc[5] = WMMA_F16(a1, b1, acc[5]);
  acc[6] = WMMA_F16(a1, b2, acc[6]);
  acc[7] = WMMA_F16(a1, b3, acc[7]);

  int mro = (lane >> 4) * 8;
#pragma unroll
  for (int hh = 0; hh < 2; ++hh){
    int mb = mbase + hh * 16;
#pragma unroll
    for (int t = 0; t < 4; ++t){
      int col = nbase0 + t * 16 + lrow;
      if (col >= N) continue;
      float bv = bias ? bias[col] : 0.f;
#pragma unroll
      for (int r = 0; r < 8; ++r){
        int rowi = mb + mro + r;
        if (rowi >= M) continue;
        float v = (acc[hh * 4 + t][r] + bv) * scale;
        if (relu) v = fmaxf(v, 0.f);
        if (C)  C [(size_t)rowi * N + col] = v;
        if (Ch) Ch[(size_t)rowi * N + col] = (half_t)v;
      }
    }
  }
}

// ---------------------------------------------------------------- V transpose
// vt[c][tok] = qkvh[tok][1024 + c]; pad tokens 3600..3615 with zeros
__global__ void k_vt(const half_t* __restrict__ qkvh, half_t* __restrict__ vt){
  int c = blockIdx.x; // 0..511
  for (int r = threadIdx.x; r < VT_LD; r += 256){
    half_t v = (half_t)0.f;
    if (r < NTOK) v = qkvh[(size_t)r * 1536 + 1024 + c];
    vt[(size_t)c * VT_LD + r] = v;
  }
}

// ---------------------------------------------------------------- flash attention
// one wave per (16-query tile, head); block-causal mask via stepOf.
// No online max (scores bounded; softmax shift-invariant; f32 exp safe);
// per-lane partial row sums, single reduction at the end.
// V fragments issued right after the S-WMMAs so they are in flight during
// the exp/mask/LDS-relayout phase.
__global__ __launch_bounds__(32, 1)
void k_attn(const half_t* __restrict__ qkvh, const half_t* __restrict__ vt,
            half_t* __restrict__ oh)
{
  const float SC = 0.125f; // 1/sqrt(64)
  int qt   = blockIdx.x;          // 0..224
  int h    = blockIdx.y;          // 0..7
  int lane = threadIdx.x;         // 32
  int qbase = qt * 16;
  int row  = lane & 15;
  int khf  = (lane >> 4) * 8;
  int ncol = lane & 15;
  int mro  = (lane >> 4) * 8;

  __shared__ half_t pbuf[16 * 32];

  // Q fragments (A-layout, 16 queries x 64 features, two K=32 chunks)
  const half_t* qr = qkvh + (size_t)(qbase + row) * 1536 + h * 64;
  v16h qa0 = ld16(qr + khf,      qr + khf + 16);
  v16h qa1 = ld16(qr + 32 + khf, qr + 48 + khf);

  int smax = 0;
#pragma unroll
  for (int i = 0; i < 16; ++i){ int s = stepOf(qbase + i); smax = s > smax ? s : smax; }

  float psum[8];
  v8f oacc[4] = {};
#pragma unroll
  for (int r = 0; r < 8; ++r) psum[r] = 0.f;

  for (int kb = 0; kb < 3616; kb += 32){
    // skip key blocks whose minimum step exceeds the tile's max step
    int ms = stepOf(kb < NTOK ? kb : NTOK - 1);
    if (kb < 3000 && kb + 31 >= 3000) ms = 0;
    if (kb < 3300 && kb + 31 >= 3300) ms = 0;
    if (ms > smax) continue;

    int key0 = kb + row;      if (key0 > NTOK - 1) key0 = NTOK - 1;
    int key1 = kb + 16 + row; if (key1 > NTOK - 1) key1 = NTOK - 1;
    const half_t* kr0 = qkvh + (size_t)key0 * 1536 + 512 + h * 64;
    const half_t* kr1 = qkvh + (size_t)key1 * 1536 + 512 + h * 64;

    v8f s0 = {}, s1 = {};
    s0 = WMMA_F16(qa0, ld16(kr0 + khf,      kr0 + khf + 16), s0);
    s0 = WMMA_F16(qa1, ld16(kr0 + 32 + khf, kr0 + 48 + khf), s0);
    s1 = WMMA_F16(qa0, ld16(kr1 + khf,      kr1 + khf + 16), s1);
    s1 = WMMA_F16(qa1, ld16(kr1 + 32 + khf, kr1 + 48 + khf), s1);

    // V fragments: independent of scores -> issue now, consume after softmax
    const half_t* vr0 = vt + (size_t)(h * 64 +  0 + row) * VT_LD + kb + khf;
    const half_t* vr1 = vt + (size_t)(h * 64 + 16 + row) * VT_LD + kb + khf;
    const half_t* vr2 = vt + (size_t)(h * 64 + 32 + row) * VT_LD + kb + khf;
    const half_t* vr3 = vt + (size_t)(h * 64 + 48 + row) * VT_LD + kb + khf;
    v16h vb0 = ld16(vr0, vr0 + 16);
    v16h vb1 = ld16(vr1, vr1 + 16);
    v16h vb2 = ld16(vr2, vr2 + 16);
    v16h vb3 = ld16(vr3, vr3 + 16);

    int k0 = kb + ncol, k1 = kb + 16 + ncol;
    bool kv0 = k0 < NTOK, kv1 = k1 < NTOK;
    int ks0 = stepOf(kv0 ? k0 : NTOK - 1);
    int ks1 = stepOf(kv1 ? k1 : NTOK - 1);

#pragma unroll
    for (int r = 0; r < 8; ++r){
      int qs = stepOf(qbase + mro + r);
      float p0 = (kv0 && ks0 <= qs) ? __expf(s0[r] * SC) : 0.f;
      float p1 = (kv1 && ks1 <= qs) ? __expf(s1[r] * SC) : 0.f;
      psum[r] += p0 + p1;
      pbuf[(mro + r) * 32 + ncol]      = (half_t)p0;  // relayout P: C-layout -> LDS
      pbuf[(mro + r) * 32 + 16 + ncol] = (half_t)p1;
    }
    __syncthreads();
    const half_t* pp = &pbuf[row * 32 + khf];
    v16h pa = ld16(pp, pp + 16);                      // P as A-fragment (16x32)
    __syncthreads();

    oacc[0] = WMMA_F16(pa, vb0, oacc[0]);
    oacc[1] = WMMA_F16(pa, vb1, oacc[1]);
    oacc[2] = WMMA_F16(pa, vb2, oacc[2]);
    oacc[3] = WMMA_F16(pa, vb3, oacc[3]);
  }

  // row-sum reduction across the 16 lanes holding each row, then normalize
  float linv[8];
#pragma unroll
  for (int r = 0; r < 8; ++r){
    float l = psum[r];
#pragma unroll
    for (int off = 1; off < 16; off <<= 1) l += __shfl_xor(l, off, 32);
    linv[r] = 1.f / l;
  }
#pragma unroll
  for (int f = 0; f < 4; ++f)
#pragma unroll
    for (int r = 0; r < 8; ++r){
      int qrow = qbase + mro + r;
      oh[(size_t)qrow * DMODEL + h * 64 + f * 16 + ncol] = (half_t)(oacc[f][r] * linv[r]);
    }
}

// ---------------------------------------------------------------- residual + LayerNorm
__global__ void k_addln(const float* __restrict__ xin, const float* __restrict__ proj,
                        const float* __restrict__ g, const float* __restrict__ b,
                        float* __restrict__ xout, half_t* __restrict__ xh)
{
  int row = blockIdx.x, tid = threadIdx.x; // 256 threads, 2 elems each
  __shared__ float sred[8];
  const float* xi = xin  + (size_t)row * DMODEL;
  const float* pr = proj + (size_t)row * DMODEL;
  float v0 = xi[tid]       + pr[tid];
  float v1 = xi[tid + 256] + pr[tid + 256];
  float s = v0 + v1;
#pragma unroll
  for (int off = 16; off >= 1; off >>= 1) s += __shfl_xor(s, off, 32);
  int w = tid >> 5;
  if ((tid & 31) == 0) sred[w] = s;
  __syncthreads();
  float tot = 0.f;
#pragma unroll
  for (int i = 0; i < 8; ++i) tot += sred[i];
  float mean = tot * (1.f / 512.f);
  float d0 = v0 - mean, d1 = v1 - mean;
  float q = d0 * d0 + d1 * d1;
#pragma unroll
  for (int off = 16; off >= 1; off >>= 1) q += __shfl_xor(q, off, 32);
  __syncthreads();
  if ((tid & 31) == 0) sred[w] = q;
  __syncthreads();
  float tv = 0.f;
#pragma unroll
  for (int i = 0; i < 8; ++i) tv += sred[i];
  float rstd = rsqrtf(tv * (1.f / 512.f) + 1e-5f);
  float o0 = d0 * rstd * g[tid]       + b[tid];
  float o1 = d1 * rstd * g[tid + 256] + b[tid + 256];
  xout[(size_t)row * DMODEL + tid]       = o0;
  xout[(size_t)row * DMODEL + tid + 256] = o1;
  xh  [(size_t)row * DMODEL + tid]       = (half_t)o0;
  xh  [(size_t)row * DMODEL + tid + 256] = (half_t)o1;
}

// ================================================================ host
extern "C" void kernel_launch(void* const* d_in, const int* in_sizes, int n_in,
                              void* d_out, int out_size, void* d_ws, size_t ws_size,
                              hipStream_t stream)
{
  const float* player_xs   = (const float*)d_in[0];
  const float* player_ys   = (const float*)d_in[1];
  const float* player_hoop = (const float*)d_in[2];
  const float* ball_xs  = (const float*)d_in[3];
  const float* ball_ys  = (const float*)d_in[4];
  const float* ball_zs  = (const float*)d_in[5];
  const float* emb_table= (const float*)d_in[6];
  const float* ball_emb = (const float*)d_in[7];
  const float* cls_emb  = (const float*)d_in[8];
  const float* pW0=(const float*)d_in[9];  const float* pb0=(const float*)d_in[10];
  const float* pW1=(const float*)d_in[11]; const float* pb1=(const float*)d_in[12];
  const float* pW2=(const float*)d_in[13]; const float* pb2=(const float*)d_in[14];
  const float* bW0=(const float*)d_in[15]; const float* bb0=(const float*)d_in[16];
  const float* bW1=(const float*)d_in[17]; const float* bb1=(const float*)d_in[18];
  const float* bW2=(const float*)d_in[19]; const float* bb2=(const float*)d_in[20];
  const float* Wqkv=(const float*)d_in[21];const float* bqkv=(const float*)d_in[22];
  const float* Wo=(const float*)d_in[23];  const float* bo=(const float*)d_in[24];
  const float* W1f=(const float*)d_in[25]; const float* b1f=(const float*)d_in[26];
  const float* W2f=(const float*)d_in[27]; const float* b2f=(const float*)d_in[28];
  const float* g1=(const float*)d_in[29];  const float* be1=(const float*)d_in[30];
  const float* g2=(const float*)d_in[31];  const float* be2=(const float*)d_in[32];
  const float* cpW=(const float*)d_in[33]; const float* cpb=(const float*)d_in[34];
  const float* cbW=(const float*)d_in[35]; const float* cbb=(const float*)d_in[36];
  const float* csW=(const float*)d_in[37]; const float* csb=(const float*)d_in[38];
  const int*   player_idxs=(const int*)d_in[39];
  float* out = (float*)d_out;
  (void)in_sizes; (void)n_in; (void)out_size; (void)ws_size;

  char* ws = (char*)d_ws;
  size_t off = 0;
  auto balloc = [&](size_t bytes)->char*{
    char* p = ws + off;
    off = (off + bytes + 255) & ~(size_t)255;
    return p;
  };
  half_t* whqkv = (half_t*)balloc((size_t)4*1536*512*2);
  half_t* whwo  = (half_t*)balloc((size_t)4*512*512*2);
  half_t* whw1  = (half_t*)balloc((size_t)4*2048*512*2);
  half_t* whw2  = (half_t*)balloc((size_t)4*512*2048*2);
  half_t* whcp  = (half_t*)balloc((size_t)121*512*2);
  half_t* whcb  = (half_t*)balloc((size_t)1331*512*2);
  half_t* whcs  = (half_t*)balloc((size_t)10*512*2);
  half_t* whp0  = (half_t*)balloc((size_t)128*32*2);
  half_t* whb0  = (half_t*)balloc((size_t)128*32*2);
  half_t* whp1  = (half_t*)balloc((size_t)256*128*2);
  half_t* whb1  = (half_t*)balloc((size_t)256*128*2);
  half_t* whp2  = (half_t*)balloc((size_t)512*256*2);
  half_t* whb2  = (half_t*)balloc((size_t)512*256*2);
  float*  x     = (float*) balloc((size_t)NTOK*DMODEL*4);
  half_t* xh    = (half_t*)balloc((size_t)NTOK*DMODEL*2);
  half_t* qkvh  = (half_t*)balloc((size_t)NTOK*1536*2);
  half_t* vt    = (half_t*)balloc((size_t)512*VT_LD*2);
  half_t* ohb   = (half_t*)balloc((size_t)NTOK*DMODEL*2);
  float*  proj  = (float*) balloc((size_t)NTOK*DMODEL*4);
  half_t* ffh   = (half_t*)balloc((size_t)NTOK*2048*2);
  half_t* inp   = (half_t*)balloc((size_t)NTOK*32*2);
  half_t* h0h   = (half_t*)balloc((size_t)NTOK*128*2);
  half_t* h1h   = (half_t*)balloc((size_t)NTOK*256*2);

  auto cvt = [&](const float* s, half_t* d, int n){
    k_cvt<<<dim3((n + 255) / 256), dim3(256), 0, stream>>>(s, d, n);
  };
  cvt(Wqkv, whqkv, 4*1536*512);
  cvt(Wo,   whwo,  4*512*512);
  cvt(W1f,  whw1,  4*2048*512);
  cvt(W2f,  whw2,  4*512*2048);
  cvt(cpW,  whcp,  121*512);
  cvt(cbW,  whcb,  1331*512);
  cvt(csW,  whcs,  10*512);
  k_cvt_pad<<<dim3((128*32+255)/256), dim3(256), 0, stream>>>(pW0, whp0, 128, 23, 32);
  k_cvt_pad<<<dim3((128*32+255)/256), dim3(256), 0, stream>>>(bW0, whb0, 128, 23, 32);
  cvt(pW1, whp1, 256*128);
  cvt(bW1, whb1, 256*128);
  cvt(pW2, whp2, 512*256);
  cvt(bW2, whb2, 512*256);

  auto gemm = [&](const half_t* A, const half_t* B, const float* bias,
                  float* C, half_t* Ch, int M, int N, int K, int relu, float scale){
    k_gemm<<<dim3((N + 127) / 128, (M + 127) / 128), dim3(256), 0, stream>>>(
        A, B, bias, C, Ch, M, N, K, relu, scale);
  };

  // ------- token MLPs via WMMA (player rows 0..2999, ball rows 3000..3299)
  k_build_in<<<dim3((3300*32 + 255)/256), dim3(256), 0, stream>>>(
      player_xs, player_ys, player_hoop, ball_xs, ball_ys, ball_zs,
      emb_table, ball_emb, player_idxs, inp);
  gemm(inp,             whp0, pb0, nullptr, h0h,            3000, 128, 32, 1, 1.f);
  gemm(inp + 3000*32,   whb0, bb0, nullptr, h0h + 3000*128,  300, 128, 32, 1, 1.f);
  gemm(h0h,             whp1, pb1, nullptr, h1h,            3000, 256, 128, 1, 1.f);
  gemm(h0h + 3000*128,  whb1, bb1, nullptr, h1h + 3000*256,  300, 256, 128, 1, 1.f);
  gemm(h1h,             whp2, pb2, x,            xh,            3000, 512, 256, 0, SQRT_D);
  gemm(h1h + 3000*256,  whb2, bb2, x + 3000*512, xh + 3000*512,  300, 512, 256, 0, SQRT_D);
  k_cls<<<dim3((300*DMODEL + 255)/256), dim3(256), 0, stream>>>(cls_emb, x, xh);

  // ------- transformer layers
  for (int l = 0; l < 4; ++l){
    gemm(xh, whqkv + (size_t)l*1536*512, bqkv + (size_t)l*1536,
         nullptr, qkvh, NTOK, 1536, 512, 0, 1.f);
    k_vt<<<dim3(512), dim3(256), 0, stream>>>(qkvh, vt);
    k_attn<<<dim3(NTOK/16, 8), dim3(32), 0, stream>>>(qkvh, vt, ohb);
    gemm(ohb, whwo + (size_t)l*512*512, bo + (size_t)l*512,
         proj, nullptr, NTOK, 512, 512, 0, 1.f);
    k_addln<<<dim3(NTOK), dim3(256), 0, stream>>>(x, proj, g1 + (size_t)l*512, be1 + (size_t)l*512, x, xh);
    gemm(xh, whw1 + (size_t)l*2048*512, b1f + (size_t)l*2048,
         nullptr, ffh, NTOK, 2048, 512, 1, 1.f);
    gemm(ffh, whw2 + (size_t)l*512*2048, b2f + (size_t)l*512,
         proj, nullptr, NTOK, 512, 2048, 0, 1.f);
    k_addln<<<dim3(NTOK), dim3(256), 0, stream>>>(x, proj, g2 + (size_t)l*512, be2 + (size_t)l*512, x, xh);
  }

  // ------- classifier heads
  gemm(xh, whcp, cpb, out,                           nullptr, NTOK, 121,  512, 0, 1.f);
  gemm(xh, whcb, cbb, out + (size_t)NTOK*121,        nullptr, NTOK, 1331, 512, 0, 1.f);
  gemm(xh, whcs, csb, out + (size_t)NTOK*(121+1331), nullptr, NTOK, 10,   512, 0, 1.f);
}
